// TGT_33165737460156
// MI455X (gfx1250) — matrix operand order repaired and checked by reading, not compiled, not run on hardware
//
#include <hip/hip_runtime.h>

typedef __attribute__((ext_vector_type(16))) _Float16 v16h;
typedef __attribute__((ext_vector_type(8)))  _Float16 v8h;
typedef __attribute__((ext_vector_type(8)))  float    v8f;

// ---------------- problem constants ----------------
#define NN      1024            // N edges in batch
#define NODES3  3072            // 3*N
#define KNB     20              // neighbors per node
#define DM      172             // D_MEM = D_EMB = D_EDGE
#define DT      100             // D_TIME
#define DMSG    616             // 2*DM + DM + DT
#define DIN     688             // D_INNER
// padded dims (K padded to mult of 32, N/M to mult of 16)
#define KP_MSG  640             // 616 -> 640
#define KP_MEM  192             // 172 -> 192
#define KP_QIN  288             // 272 -> 288
#define KP_KV   448             // 444 -> 448
#define KP_FFN  704             // 688 -> 704
#define NP_GRU  528             // 3*172=516 -> 528
#define NP_HD   176             // 172 -> 176
#define NP_FFN  704             // 688 -> 704

static __device__ __forceinline__ float sigmoidf_(float x){ return 1.0f/(1.0f+__expf(-x)); }

// ---------------- WMMA GEMM: C = A(MxKp,f16) * Bt(NxKp,f16)^T  ----------------
// Bt holds B transposed (N rows of length Kp). Fragment layouts follow the
// CDNA5 wave32 VGPR maps (CDNA5 ISA): A 16x32 f16, B 32x16 f16, C/D 16x16 f32.
__global__ void __launch_bounds__(256)
wmma_gemm(const _Float16* __restrict__ A, const _Float16* __restrict__ Bt,
          void* __restrict__ C, int M, int N, int Kp,
          const float* __restrict__ bias, int n_valid, int relu, int store_half)
{
    const int lane = threadIdx.x & 31;
    const int wave = threadIdx.x >> 5;
    const int Mt = M >> 4, Nt = N >> 4;
    long t = (long)blockIdx.x * 8 + wave;
    if (t >= (long)Mt * Nt) return;
    const int tm = (int)(t % Mt);          // consecutive waves share tn -> B frag reuse in L0
    const int tn = (int)(t / Mt);

    // A lane map: lanes 0-15 hold K {0..7,16..23}, lanes 16-31 hold K {8..15,24..31}
    const _Float16* ap = A + (size_t)(tm*16 + (lane & 15)) * Kp + ((lane >= 16) ? 8 : 0);
    // B lane map: lanes 0-15 hold K {0..7,16..23}, lanes 16-31 hold K {8..15,24..31}
    const _Float16* bp = Bt + (size_t)(tn*16 + (lane & 15)) * Kp + ((lane >= 16) ? 8 : 0);

    v8f acc = {0.f,0.f,0.f,0.f,0.f,0.f,0.f,0.f};
    for (int k = 0; k < Kp; k += 32) {
        __builtin_prefetch(ap + k + 64, 0, 1);   // global_prefetch_b8 next K tile
        __builtin_prefetch(bp + k + 64, 0, 1);
        v8h a0 = *(const v8h*)(ap + k);
        v8h a1 = *(const v8h*)(ap + k + 16);
        v8h b0 = *(const v8h*)(bp + k);
        v8h b1 = *(const v8h*)(bp + k + 16);
        v16h a, b;
#pragma unroll
        for (int j = 0; j < 8; ++j) { a[j] = a0[j]; a[8+j] = a1[j]; b[j] = b0[j]; b[8+j] = b1[j]; }
        acc = __builtin_amdgcn_wmma_f32_16x16x32_f16(false, a, false, b, (short)0, acc, false, false);
    }

    const int col   = tn*16 + (lane & 15);
    const int rbase = tm*16 + ((lane >= 16) ? 8 : 0);
#pragma unroll
    for (int r = 0; r < 8; ++r) {
        int row = rbase + r;
        float v = acc[r];
        if (col < n_valid) {
            if (bias) v += bias[col];
            if (relu) v = fmaxf(v, 0.f);
        } else {
            v = 0.f;
        }
        if (store_half) ((_Float16*)C)[(size_t)row * N + col] = (_Float16)v;
        else            ((float*)C)  [(size_t)row * N + col] = v;
    }
}

// ---------------- packing / elementwise kernels ----------------
__global__ void copy_f32(const float* __restrict__ s, float* __restrict__ d, long n){
    for (long i=(long)blockIdx.x*blockDim.x+threadIdx.x; i<n; i+=(long)gridDim.x*blockDim.x) d[i]=s[i];
}

// Wt[n*Kp+k] = W[k*N+n] (f32 -> f16, zero-padded)
__global__ void pack_wt(const float* __restrict__ W, _Float16* __restrict__ Wt,
                        int Kd, int N, int Kp, int Np){
    long tot=(long)Np*Kp;
    for (long i=(long)blockIdx.x*blockDim.x+threadIdx.x; i<tot; i+=(long)gridDim.x*blockDim.x){
        int n=(int)(i/Kp), k=(int)(i%Kp);
        float v = (n<N && k<Kd) ? W[(size_t)k*N+n] : 0.f;
        Wt[i]=(_Float16)v;
    }
}

// x for the 4 update-GRUs: 4096 x KP_MSG  [memA | memB | ef | time_enc | 0pad]
__global__ void pack_upd_x(const float* __restrict__ mem, const float* __restrict__ lu,
                           const float* __restrict__ ef, const float* __restrict__ et,
                           const float* __restrict__ tw, const float* __restrict__ tb,
                           const int* __restrict__ src, const int* __restrict__ dst,
                           const int* __restrict__ neg, const int* __restrict__ eidx,
                           _Float16* __restrict__ x){
    long tot = 4L*NN*KP_MSG;
    for (long e=(long)blockIdx.x*blockDim.x+threadIdx.x; e<tot; e+=(long)gridDim.x*blockDim.x){
        int r=(int)(e/KP_MSG), c=(int)(e%KP_MSG);
        int g=r>>10, i=r&1023;
        int na,nb,ntd;
        if      (g==0){ na=src[i]; nb=dst[i]; ntd=na; }
        else if (g==1){ na=dst[i]; nb=src[i]; ntd=na; }
        else if (g==2){ na=src[i]; nb=neg[i]; ntd=na; }
        else          { na=neg[i]; nb=src[i]; ntd=na; }
        float v=0.f;
        if      (c<DM)        v = mem[(size_t)na*DM + c];
        else if (c<2*DM)      v = mem[(size_t)nb*DM + (c-DM)];
        else if (c<2*DM+DM)   v = ef[(size_t)eidx[i]*DM + (c-2*DM)];
        else if (c<DMSG){ int j=c-3*DM; float dt=et[i]-lu[ntd]; v=__cosf(dt*tw[j]+tb[j]); }
        x[e]=(_Float16)v;
    }
}

// h rows for GRUs 1,2,4: [src_mem ; dest_mem ; neg_mem] -> 3072 x KP_MEM f16
__global__ void pack_hcat(const float* __restrict__ mem, const int* __restrict__ src,
                          const int* __restrict__ dst, const int* __restrict__ neg,
                          _Float16* __restrict__ h){
    long tot=(long)NODES3*KP_MEM;
    for (long e=(long)blockIdx.x*blockDim.x+threadIdx.x; e<tot; e+=(long)gridDim.x*blockDim.x){
        int r=(int)(e/KP_MEM), c=(int)(e%KP_MEM);
        int g=r>>10, i=r&1023;
        int nd = (g==0)?src[i]:((g==1)?dst[i]:neg[i]);
        h[e] = (c<DM) ? (_Float16)mem[(size_t)nd*DM+c] : (_Float16)0.f;
    }
}

// GRU gate combine (gi stride NP_GRU); h gathered via hidx or direct
__global__ void gru_combine(const float* __restrict__ gi, const float* __restrict__ gh,
                            const float* __restrict__ hsrc, const int* __restrict__ hidx,
                            const float* __restrict__ bih, const float* __restrict__ bhh,
                            float* __restrict__ out, _Float16* __restrict__ outh, int rows){
    long tot=(long)rows*KP_MEM;
    for (long e=(long)blockIdx.x*blockDim.x+threadIdx.x; e<tot; e+=(long)gridDim.x*blockDim.x){
        int r=(int)(e/KP_MEM), c=(int)(e%KP_MEM);
        if (c>=DM){ if (outh) outh[(size_t)r*KP_MEM+c]=(_Float16)0.f; continue; }
        size_t gb=(size_t)r*NP_GRU;
        float ir=gi[gb+c]+bih[c],       iz=gi[gb+DM+c]+bih[DM+c],   ig=gi[gb+2*DM+c]+bih[2*DM+c];
        float hr=gh[gb+c]+bhh[c],       hz=gh[gb+DM+c]+bhh[DM+c],   hg=gh[gb+2*DM+c]+bhh[2*DM+c];
        float h  = hidx ? hsrc[(size_t)hidx[r]*DM+c] : hsrc[(size_t)r*DM+c];
        float rg = sigmoidf_(ir+hr), z = sigmoidf_(iz+hz);
        float g  = tanhf(ig + rg*hg);
        float u  = (1.f-z)*g + z*h;
        out[(size_t)r*DM+c]=u;
        if (outh) outh[(size_t)r*KP_MEM+c]=(_Float16)u;
    }
}

__global__ void scatter_rows(float* __restrict__ mem, const int* __restrict__ idx,
                             const float* __restrict__ rows, int n){
    long tot=(long)n*DM;
    for (long e=(long)blockIdx.x*blockDim.x+threadIdx.x; e<tot; e+=(long)gridDim.x*blockDim.x){
        int r=(int)(e/DM), c=(int)(e%DM);
        mem[(size_t)idx[r]*DM+c] = rows[(size_t)r*DM+c];
    }
}

// q_in: 3072 x KP_QIN  [mem2[node] | cos(tb) | 0]
__global__ void pack_q(const float* __restrict__ mem2, const int* __restrict__ src,
                       const int* __restrict__ dst, const int* __restrict__ neg,
                       const float* __restrict__ tb, _Float16* __restrict__ q){
    long tot=(long)NODES3*KP_QIN;
    for (long e=(long)blockIdx.x*blockDim.x+threadIdx.x; e<tot; e+=(long)gridDim.x*blockDim.x){
        int r=(int)(e/KP_QIN), c=(int)(e%KP_QIN);
        int g=r>>10, i=r&1023;
        int nd=(g==0)?src[i]:((g==1)?dst[i]:neg[i]);
        float v=0.f;
        if (c<DM) v=mem2[(size_t)nd*DM+c];
        else if (c<DM+DT) v=__cosf(tb[c-DM]);
        q[e]=(_Float16)v;
    }
}

// kv_in: 61440 x KP_KV  [mem2[nb] | ef[edge] | time_enc | 0]
__global__ void pack_kv(const float* __restrict__ mem2, const float* __restrict__ ef,
                        const float* __restrict__ et, const float* __restrict__ nt,
                        const float* __restrict__ tw, const float* __restrict__ tb,
                        const int* __restrict__ nbi, const int* __restrict__ eidx,
                        _Float16* __restrict__ kv){
    long tot=(long)NODES3*KNB*KP_KV;
    for (long e=(long)blockIdx.x*blockDim.x+threadIdx.x; e<tot; e+=(long)gridDim.x*blockDim.x){
        int r=(int)(e/KP_KV), c=(int)(e%KP_KV);
        float v=0.f;
        if      (c<DM)      v = mem2[(size_t)nbi[r]*DM+c];
        else if (c<2*DM)    v = ef[(size_t)eidx[r]*DM+(c-DM)];
        else if (c<2*DM+DT){
            int b=r/KNB, j=c-2*DM;
            float dt = et[b & 1023] - nt[r];
            v=__cosf(dt*tw[j]+tb[j]);
        }
        kv[e]=(_Float16)v;
    }
}

// f32 (M x srcw) -> f16 (M x Kp) zero-padded
__global__ void pack_rows(const float* __restrict__ s, _Float16* __restrict__ d,
                          int M, int srcw, int Kp){
    long tot=(long)M*Kp;
    for (long e=(long)blockIdx.x*blockDim.x+threadIdx.x; e<tot; e+=(long)gridDim.x*blockDim.x){
        int r=(int)(e/Kp), c=(int)(e%Kp);
        d[e] = (c<srcw) ? (_Float16)s[(size_t)r*srcw+c] : (_Float16)0.f;
    }
}

// per-query attention: softmax over 20 neighbors, 2 heads; LDS-resident scores
__global__ void __launch_bounds__(192)
attn_kernel(const float* __restrict__ Q, const _Float16* __restrict__ Kh,
            const _Float16* __restrict__ Vh, const int* __restrict__ nbi,
            _Float16* __restrict__ ctx){
    __shared__ float sQ[DM];
    __shared__ float sc[2][KNB];
    int b=blockIdx.x, tid=threadIdx.x;
    if (tid<DM) sQ[tid]=Q[(size_t)b*NP_HD+tid];
    __syncthreads();
    if (tid<2*KNB){
        int h=tid/KNB, n=tid%KNB;
        const _Float16* kr = Kh + ((size_t)b*KNB+n)*NP_HD + h*86;
        const float* qh = sQ + h*86;
        float s=0.f;
        for (int d=0; d<86; ++d) s += qh[d]*(float)kr[d];
        s *= 0.10783277320343841f;            // 1/sqrt(86)
        if (nbi[b*KNB+n]==0) s=-1.0e9f;
        sc[h][n]=s;
    }
    __syncthreads();
    if (tid<2){
        float m=-3.4e38f;
        for (int n=0;n<KNB;++n) m=fmaxf(m,sc[tid][n]);
        float sum=0.f;
        for (int n=0;n<KNB;++n){ float e=__expf(sc[tid][n]-m); sc[tid][n]=e; sum+=e; }
        float inv=1.f/sum;
        for (int n=0;n<KNB;++n) sc[tid][n]*=inv;
    }
    __syncthreads();
    float v=0.f;
    if (tid<DM){
        int h=tid/86;
        for (int n=0;n<KNB;++n) v += sc[h][n]*(float)Vh[((size_t)b*KNB+n)*NP_HD+tid];
    }
    if (tid<KP_MEM) ctx[(size_t)b*KP_MEM+tid]=(_Float16)v;
}

// emb = attn_out + (ffn2 + b2); probs = sigmoid(concat @ mlp_W + b)
__global__ void final_mlp(const float* __restrict__ ao, const float* __restrict__ h2,
                          const float* __restrict__ b2, const float* __restrict__ mw,
                          const float* __restrict__ mb, float* __restrict__ out){
    int i=blockIdx.x*blockDim.x+threadIdx.x;
    if (i>=NN) return;
    float accp=mb[0], accn=mb[0];
    for (int c=0;c<DM;++c){
        float es=ao[(size_t)i*NP_HD+c]        + h2[(size_t)i*NP_HD+c]        + b2[c];
        float ed=ao[(size_t)(NN+i)*NP_HD+c]   + h2[(size_t)(NN+i)*NP_HD+c]   + b2[c];
        float en=ao[(size_t)(2*NN+i)*NP_HD+c] + h2[(size_t)(2*NN+i)*NP_HD+c] + b2[c];
        float w0=mw[c], w1=mw[DM+c];
        accp += w0*es + w1*ed;
        accn += w0*es + w1*en;
    }
    out[i]     = sigmoidf_(accp);
    out[NN+i]  = sigmoidf_(accn);
}

// ---------------- host launcher ----------------
extern "C" void kernel_launch(void* const* d_in, const int* in_sizes, int n_in,
                              void* d_out, int out_size, void* d_ws, size_t ws_size,
                              hipStream_t stream) {
    (void)in_sizes; (void)n_in; (void)out_size; (void)ws_size;
    const float* memory   = (const float*)d_in[0];
    const float* last_upd = (const float*)d_in[1];
    const float* ef_tab   = (const float*)d_in[2];
    const float* edge_t   = (const float*)d_in[3];
    const float* nbr_t    = (const float*)d_in[4];
    const float* time_w   = (const float*)d_in[5];
    const float* time_b   = (const float*)d_in[6];
    const float* upd_Wih  = (const float*)d_in[7];
    const float* upd_Whh  = (const float*)d_in[8];
    const float* upd_bih  = (const float*)d_in[9];
    const float* upd_bhh  = (const float*)d_in[10];
    // d_in[11..14] prop_* : dead code w.r.t. outputs (memory not returned)
    const float* Wq       = (const float*)d_in[15];
    const float* Wk       = (const float*)d_in[16];
    const float* Wv       = (const float*)d_in[17];
    const float* Wo       = (const float*)d_in[18];
    const float* ffn_W1   = (const float*)d_in[19];
    const float* ffn_b1   = (const float*)d_in[20];
    const float* ffn_W2   = (const float*)d_in[21];
    const float* ffn_b2   = (const float*)d_in[22];
    const float* mlp_W    = (const float*)d_in[23];
    const float* mlp_b    = (const float*)d_in[24];
    const int*   src      = (const int*)d_in[25];
    const int*   dst      = (const int*)d_in[26];
    const int*   neg      = (const int*)d_in[27];
    const int*   e_sd     = (const int*)d_in[28];
    const int*   nbi      = (const int*)d_in[29];
    const int*   eidx     = (const int*)d_in[30];
    float* out = (float*)d_out;

    char* ws=(char*)d_ws; size_t off=0;
    auto alloc=[&](size_t bytes)->void*{ void* p=ws+off; off=(off+bytes+255)&~(size_t)255; return p; };

    float*    mem2     = (float*)   alloc((size_t)30000*DM*4);
    _Float16* wtUih    = (_Float16*)alloc((size_t)NP_GRU*KP_MSG*2);
    _Float16* wtUhh    = (_Float16*)alloc((size_t)NP_GRU*KP_MEM*2);
    _Float16* wtQ      = (_Float16*)alloc((size_t)NP_HD*KP_QIN*2);
    _Float16* wtK      = (_Float16*)alloc((size_t)NP_HD*KP_KV*2);
    _Float16* wtV      = (_Float16*)alloc((size_t)NP_HD*KP_KV*2);
    _Float16* wtO      = (_Float16*)alloc((size_t)NP_HD*KP_MEM*2);
    _Float16* wtF1     = (_Float16*)alloc((size_t)NP_FFN*KP_MEM*2);
    _Float16* wtF2     = (_Float16*)alloc((size_t)NP_HD*KP_FFN*2);
    _Float16* xcat     = (_Float16*)alloc((size_t)4*NN*KP_MSG*2);
    _Float16* hcat     = (_Float16*)alloc((size_t)NODES3*KP_MEM*2);
    float*    gi       = (float*)   alloc((size_t)4*NN*NP_GRU*4);
    float*    gh       = (float*)   alloc((size_t)NODES3*NP_GRU*4);
    float*    gh3      = (float*)   alloc((size_t)NN*NP_GRU*4);
    float*    u_srcA   = (float*)   alloc((size_t)NN*DM*4);
    _Float16* u_srcA16 = (_Float16*)alloc((size_t)NN*KP_MEM*2);
    float*    u_dest   = (float*)   alloc((size_t)NN*DM*4);
    float*    u_neg    = (float*)   alloc((size_t)NN*DM*4);
    float*    u_src    = (float*)   alloc((size_t)NN*DM*4);
    _Float16* qpack    = (_Float16*)alloc((size_t)NODES3*KP_QIN*2);
    _Float16* kvpack   = (_Float16*)alloc((size_t)NODES3*KNB*KP_KV*2);
    float*    Qf32     = (float*)   alloc((size_t)NODES3*NP_HD*4);
    _Float16* Kh16     = (_Float16*)alloc((size_t)NODES3*KNB*NP_HD*2);
    _Float16* Vh16     = (_Float16*)alloc((size_t)NODES3*KNB*NP_HD*2);
    _Float16* ctx16    = (_Float16*)alloc((size_t)NODES3*KP_MEM*2);
    float*    aout     = (float*)   alloc((size_t)NODES3*NP_HD*4);
    _Float16* aout16   = (_Float16*)alloc((size_t)NODES3*KP_MEM*2);
    _Float16* h1_16    = (_Float16*)alloc((size_t)NODES3*NP_FFN*2);
    float*    h2f      = (float*)   alloc((size_t)NODES3*NP_HD*4);

    auto grd=[&](long tot){ return dim3((unsigned)((tot+255)/256)); };
    auto gemm=[&](const _Float16* A,const _Float16* Bt,void* C,int M,int N,int Kp,
                  const float* bias,int nv,int relu,int sh){
        long tiles=(long)(M/16)*(N/16);
        wmma_gemm<<<dim3((unsigned)((tiles+7)/8)),256,0,stream>>>(A,Bt,C,M,N,Kp,bias,nv,relu,sh);
    };

    // weights -> f16 transposed+padded
    pack_wt<<<grd((long)NP_GRU*KP_MSG),256,0,stream>>>(upd_Wih,wtUih,DMSG,3*DM,KP_MSG,NP_GRU);
    pack_wt<<<grd((long)NP_GRU*KP_MEM),256,0,stream>>>(upd_Whh,wtUhh,DM,3*DM,KP_MEM,NP_GRU);
    pack_wt<<<grd((long)NP_HD*KP_QIN),256,0,stream>>>(Wq,wtQ,DM+DT,DM,KP_QIN,NP_HD);
    pack_wt<<<grd((long)NP_HD*KP_KV),256,0,stream>>>(Wk,wtK,2*DM+DT,DM,KP_KV,NP_HD);
    pack_wt<<<grd((long)NP_HD*KP_KV),256,0,stream>>>(Wv,wtV,2*DM+DT,DM,KP_KV,NP_HD);
    pack_wt<<<grd((long)NP_HD*KP_MEM),256,0,stream>>>(Wo,wtO,DM,DM,KP_MEM,NP_HD);
    pack_wt<<<grd((long)NP_FFN*KP_MEM),256,0,stream>>>(ffn_W1,wtF1,DM,DIN,KP_MEM,NP_FFN);
    pack_wt<<<grd((long)NP_HD*KP_FFN),256,0,stream>>>(ffn_W2,wtF2,DIN,DM,KP_FFN,NP_HD);

    // ---- update GRUs ----
    pack_upd_x<<<grd(4L*NN*KP_MSG),256,0,stream>>>(memory,last_upd,ef_tab,edge_t,time_w,time_b,
                                                   src,dst,neg,e_sd,xcat);
    pack_hcat<<<grd((long)NODES3*KP_MEM),256,0,stream>>>(memory,src,dst,neg,hcat);
    gemm(xcat,wtUih,gi,4*NN,NP_GRU,KP_MSG,nullptr,NP_GRU,0,0);
    gemm(hcat,wtUhh,gh,NODES3,NP_GRU,KP_MEM,nullptr,NP_GRU,0,0);
    gru_combine<<<grd((long)NN*KP_MEM),256,0,stream>>>(gi, gh, memory,src, upd_bih,upd_bhh,u_srcA,u_srcA16,NN);
    gru_combine<<<grd((long)NN*KP_MEM),256,0,stream>>>(gi+(size_t)NN*NP_GRU, gh+(size_t)NN*NP_GRU,
                                                       memory,dst, upd_bih,upd_bhh,u_dest,nullptr,NN);
    gru_combine<<<grd((long)NN*KP_MEM),256,0,stream>>>(gi+(size_t)3*NN*NP_GRU, gh+(size_t)2*NN*NP_GRU,
                                                       memory,neg, upd_bih,upd_bhh,u_neg,nullptr,NN);
    gemm(u_srcA16,wtUhh,gh3,NN,NP_GRU,KP_MEM,nullptr,NP_GRU,0,0);
    gru_combine<<<grd((long)NN*KP_MEM),256,0,stream>>>(gi+(size_t)2*NN*NP_GRU, gh3,
                                                       u_srcA,nullptr, upd_bih,upd_bhh,u_src,nullptr,NN);

    // ---- memory update (sequential scatters preserve src->dest->neg override order) ----
    copy_f32<<<grd(30000L*DM),256,0,stream>>>(memory,mem2,30000L*DM);
    scatter_rows<<<grd((long)NN*DM),256,0,stream>>>(mem2,src,u_src,NN);
    scatter_rows<<<grd((long)NN*DM),256,0,stream>>>(mem2,dst,u_dest,NN);
    scatter_rows<<<grd((long)NN*DM),256,0,stream>>>(mem2,neg,u_neg,NN);

    // ---- attention ----
    pack_q<<<grd((long)NODES3*KP_QIN),256,0,stream>>>(mem2,src,dst,neg,time_b,qpack);
    pack_kv<<<grd((long)NODES3*KNB*KP_KV),256,0,stream>>>(mem2,ef_tab,edge_t,nbr_t,time_w,time_b,
                                                          nbi,eidx,kvpack);
    gemm(qpack,wtQ,Qf32,NODES3,NP_HD,KP_QIN,nullptr,NP_HD,0,0);
    gemm(kvpack,wtK,Kh16,NODES3*KNB,NP_HD,KP_KV,nullptr,NP_HD,0,1);
    gemm(kvpack,wtV,Vh16,NODES3*KNB,NP_HD,KP_KV,nullptr,NP_HD,0,1);
    attn_kernel<<<dim3(NODES3),192,0,stream>>>(Qf32,Kh16,Vh16,nbi,ctx16);
    gemm(ctx16,wtO,aout,NODES3,NP_HD,KP_MEM,nullptr,NP_HD,0,0);

    // ---- FFN + residual + link probabilities ----
    pack_rows<<<grd((long)NODES3*KP_MEM),256,0,stream>>>(aout,aout16,NODES3,NP_HD,KP_MEM);
    gemm(aout16,wtF1,h1_16,NODES3,NP_FFN,KP_MEM,ffn_b1,DIN,1,1);   // bias+ReLU fused, f16 out
    gemm(h1_16,wtF2,h2f,NODES3,NP_HD,KP_FFN,nullptr,NP_HD,0,0);
    final_mlp<<<dim3(4),256,0,stream>>>(aout,h2f,ffn_b2,mlp_W,mlp_b,out);
}
